// ContinuousMamba2Model_26113401160186
// MI455X (gfx1250) — matrix operand order repaired and checked by reading, not compiled, not run on hardware
//
#include <hip/hip_runtime.h>
#include <hip/hip_bf16.h>

// ---------------- model constants ----------------
#define V_     50257
#define DM_    1024
#define NL_    4
#define BT_    2
#define T_     2048
#define DST_   128
#define DC_    4
#define HDIM_  64
#define DIN_   2048
#define NH_    32
#define CONV_  2304
#define DPROJ_ 4384
#define CHUNK_ 256
#define NC_    8
#define ROWS_  4096
#define EPS_RMS_   1.1920928955078125e-07f
#define EPS_GATED_ 1e-5f

// ---------------- WMMA plumbing ----------------
typedef __attribute__((ext_vector_type(16))) __bf16 bf16x16;
typedef __attribute__((ext_vector_type(8)))  float  f32x8;

union Frag {
  bf16x16 v;
  int4 q[2];
  unsigned short u[16];
};

__device__ __forceinline__ unsigned short f2bf(float f) {
  unsigned int u = __float_as_uint(f);
  u += 0x7fffu + ((u >> 16) & 1u);
  return (unsigned short)(u >> 16);
}
__device__ __forceinline__ float bf2f(unsigned short h) {
  return __uint_as_float(((unsigned int)h) << 16);
}

// A fragment: element (m = lane&15, k) ; k set = {half*8+0..7, 16+half*8+0..7}
__device__ __forceinline__ void load_mat_a(Frag& f, const unsigned short* base,
                                           int ld, int m0, int k0, int lane) {
  int half = lane >> 4;
  const unsigned short* p = base + (size_t)(m0 + (lane & 15)) * ld + k0 + half * 8;
  f.q[0] = *reinterpret_cast<const int4*>(p);
  f.q[1] = *reinterpret_cast<const int4*>(p + 16);
}
// B fragment from W[N,K] row-major: lane holds col n = n0+(lane&15), K = k0+{half*8+0..7, 16+half*8+0..7}
__device__ __forceinline__ void load_mat_b(Frag& f, const unsigned short* base,
                                           int ld, int n0, int k0, int lane, int nmax) {
  int half = lane >> 4;
  int n = n0 + (lane & 15);
  if (n > nmax) n = nmax;
  const unsigned short* p = base + (size_t)n * ld + k0 + half * 8;
  f.q[0] = *reinterpret_cast<const int4*>(p);
  f.q[1] = *reinterpret_cast<const int4*>(p + 16);
}
__device__ __forceinline__ f32x8 wmma_bf16(const Frag& a, const Frag& b, f32x8 c) {
  return __builtin_amdgcn_wmma_f32_16x16x32_bf16(false, a.v, false, b.v, (short)0, c,
                                                 false, false);
}
__device__ __forceinline__ f32x8 zero8() {
  f32x8 z = {0.f, 0.f, 0.f, 0.f, 0.f, 0.f, 0.f, 0.f};
  return z;
}

// ---------------- elementwise / small kernels ----------------
__global__ void f2bf_kernel(const float* __restrict__ s, unsigned short* __restrict__ d,
                            size_t n) {
  for (size_t i = (size_t)blockIdx.x * blockDim.x + threadIdx.x; i < n;
       i += (size_t)gridDim.x * blockDim.x)
    d[i] = f2bf(s[i]);
}

__global__ void embed_kernel(const int* __restrict__ x, const float* __restrict__ embed,
                             const float* __restrict__ pos, float* __restrict__ h) {
  int row = blockIdx.x;                 // b*T + t
  int t = row % T_;
  int tok = x[row];
  for (int k = 0; k < DM_ / 256; ++k) {
    int c = threadIdx.x + k * 256;
    h[(size_t)row * DM_ + c] = embed[(size_t)tok * DM_ + c] + pos[(size_t)t * DM_ + c];
  }
}

__device__ __forceinline__ float block_reduce_sum256(float v, float* red) {
  int tid = threadIdx.x;
  red[tid] = v;
  __syncthreads();
  for (int off = 128; off > 0; off >>= 1) {
    if (tid < off) red[tid] += red[tid + off];
    __syncthreads();
  }
  float r = red[0];
  __syncthreads();
  return r;
}

__global__ void rmsnorm_kernel(const float* __restrict__ x, const float* __restrict__ w,
                               unsigned short* __restrict__ out, float eps) {
  __shared__ float red[256];
  int row = blockIdx.x;
  const float* xr = x + (size_t)row * DM_;
  float g[DM_ / 256];
  float ss = 0.f;
#pragma unroll
  for (int k = 0; k < DM_ / 256; ++k) {
    int c = threadIdx.x + k * 256;
    g[k] = xr[c];
    ss += g[k] * g[k];
  }
  ss = block_reduce_sum256(ss, red);
  float s = rsqrtf(ss / (float)DM_ + eps);
#pragma unroll
  for (int k = 0; k < DM_ / 256; ++k) {
    int c = threadIdx.x + k * 256;
    out[(size_t)row * DM_ + c] = f2bf(g[k] * s * w[c]);
  }
}

__global__ void gated_norm_kernel(const float* __restrict__ y, const float* __restrict__ zx,
                                  const float* __restrict__ gw,
                                  unsigned short* __restrict__ out) {
  __shared__ float red[256];
  int row = blockIdx.x;
  const float* yr = y + (size_t)row * DIN_;
  const float* zr = zx + (size_t)row * DPROJ_;   // z = first DIN_ entries
  float g[DIN_ / 256];
  float ss = 0.f;
#pragma unroll
  for (int k = 0; k < DIN_ / 256; ++k) {
    int c = threadIdx.x + k * 256;
    float zv = zr[c];
    float sig = 1.f / (1.f + __expf(-zv));
    g[k] = yr[c] * (zv * sig);
    ss += g[k] * g[k];
  }
  ss = block_reduce_sum256(ss, red);
  float s = rsqrtf(ss / (float)DIN_ + EPS_GATED_);
#pragma unroll
  for (int k = 0; k < DIN_ / 256; ++k) {
    int c = threadIdx.x + k * 256;
    out[(size_t)row * DIN_ + c] = f2bf(g[k] * s * gw[c]);
  }
}

// causal depthwise conv + SiLU; scatter into WMMA-friendly layouts
__global__ void conv_kernel(const float* __restrict__ zx, const float* __restrict__ cw,
                            const float* __restrict__ cb, unsigned short* __restrict__ xT,
                            unsigned short* __restrict__ Bn, unsigned short* __restrict__ Bt,
                            unsigned short* __restrict__ Cn) {
  int row = blockIdx.x;
  int b = row / T_, t = row % T_;
  for (int c = threadIdx.x; c < CONV_; c += 256) {
    float acc = cb[c];
#pragma unroll
    for (int i = 0; i < DC_; ++i) {
      int tt = t - (DC_ - 1) + i;
      if (tt >= 0)
        acc += zx[((size_t)b * T_ + tt) * DPROJ_ + DIN_ + c] * cw[c * DC_ + i];
    }
    float s = acc / (1.f + __expf(-acc));
    unsigned short bv = f2bf(s);
    if (c < DIN_) {
      int hh = c >> 6, p = c & 63;
      xT[(((size_t)b * NH_ + hh) * HDIM_ + p) * T_ + t] = bv;
    } else if (c < DIN_ + DST_) {
      int n = c - DIN_;
      Bn[((size_t)b * T_ + t) * DST_ + n] = bv;
      Bt[((size_t)b * DST_ + n) * T_ + t] = bv;
    } else {
      int n = c - DIN_ - DST_;
      Cn[((size_t)b * T_ + t) * DST_ + n] = bv;
    }
  }
}

// dt = softplus(raw+bias); a = dt*A; per-chunk inclusive cumsum; derived decays
__global__ void dt_kernel(const float* __restrict__ zx, const float* __restrict__ dt_bias,
                          const float* __restrict__ A_log, float* __restrict__ dtb,
                          float* __restrict__ acb, float* __restrict__ wb,
                          float* __restrict__ eb) {
  __shared__ float sb[CHUNK_];
  int hh = blockIdx.x % NH_, b = blockIdx.x / NH_;
  int tid = threadIdx.x;
  float Ab = -__expf(A_log[hh]);
  float bias = dt_bias[hh];
  size_t obh = ((size_t)b * NH_ + hh) * T_;
  for (int c = 0; c < NC_; ++c) {
    int t = c * CHUNK_ + tid;
    float raw = zx[((size_t)b * T_ + t) * DPROJ_ + DIN_ + CONV_ + hh] + bias;
    float dt = raw > 20.f ? raw : log1pf(__expf(raw));
    float a = dt * Ab;
    sb[tid] = a;
    __syncthreads();
    for (int off = 1; off < CHUNK_; off <<= 1) {    // Hillis-Steele inclusive scan
      float v = (tid >= off) ? sb[tid - off] : 0.f;
      __syncthreads();
      sb[tid] += v;
      __syncthreads();
    }
    float ac = sb[tid];
    float alast = sb[CHUNK_ - 1];
    __syncthreads();
    dtb[obh + t] = dt;
    acb[obh + t] = ac;
    wb[obh + t] = __expf(alast - ac) * dt;   // decay_to_end * dt
    eb[obh + t] = __expf(ac);                // exp(a_cum)  (last elem = chunk_decay)
  }
}

// ---------------- generic WMMA GEMM: C[M,N] (+)= A[M,K] @ B[N,K]^T  (bf16 in, f32 out)
// per-wave 64x64 tile: 16 wmma per 16 b128 loads; K-tile prefetch via global_prefetch
__global__ __launch_bounds__(32) void wmma_gemm_kernel(
    const unsigned short* __restrict__ A, const unsigned short* __restrict__ B,
    float* __restrict__ C, int M, int N, int K, int lda, int ldb, int ldc,
    long long sA, long long sB, long long sC, int accumulate) {
  int z = blockIdx.z;
  A += (size_t)z * sA;
  B += (size_t)z * sB;
  C += (size_t)z * sC;
  int m0 = blockIdx.y * 64;
  int n0 = blockIdx.x * 64;
  int lane = threadIdx.x & 31;
  int half = lane >> 4, l15 = lane & 15;
  f32x8 acc[4][4];
#pragma unroll
  for (int i = 0; i < 4; ++i)
#pragma unroll
    for (int j = 0; j < 4; ++j) acc[i][j] = zero8();
  for (int k0 = 0; k0 < K; k0 += 32) {
    if (k0 + 64 < K) {   // prefetch next K-tile (lowers to global_prefetch_b8)
      __builtin_prefetch(A + (size_t)(m0 + lane) * lda + k0 + 64, 0, 1);
      int np = n0 + lane;
      if (np >= N) np = N - 1;
      __builtin_prefetch(B + (size_t)np * ldb + k0 + 64, 0, 1);
    }
    Frag a[4], bf[4];
#pragma unroll
    for (int fm = 0; fm < 4; ++fm) load_mat_a(a[fm], A, lda, m0 + fm * 16, k0, lane);
#pragma unroll
    for (int fn = 0; fn < 4; ++fn) load_mat_b(bf[fn], B, ldb, n0 + fn * 16, k0, lane, N - 1);
#pragma unroll
    for (int fm = 0; fm < 4; ++fm)
#pragma unroll
      for (int fn = 0; fn < 4; ++fn) acc[fm][fn] = wmma_bf16(a[fm], bf[fn], acc[fm][fn]);
  }
#pragma unroll
  for (int fm = 0; fm < 4; ++fm)
#pragma unroll
    for (int fn = 0; fn < 4; ++fn)
#pragma unroll
      for (int r = 0; r < 8; ++r) {
        int m = m0 + fm * 16 + r + 8 * half;
        int n = n0 + fn * 16 + l15;
        if (n < N) {
          size_t idx = (size_t)m * ldc + n;
          float v = acc[fm][fn][r];
          if (accumulate) v += C[idx];
          C[idx] = v;
        }
      }
}

// ---------------- SSD: intra-chunk  y = (L .* CB .* dt_j) @ x  + D*x ----------------
__global__ __launch_bounds__(256) void ssd_intra_kernel(
    const float* __restrict__ CB, const unsigned short* __restrict__ xT,
    const float* __restrict__ dtb, const float* __restrict__ acb,
    const float* __restrict__ Dpar, float* __restrict__ y) {
  __shared__ float sdt[CHUNK_], sac[CHUNK_];
  int blk = blockIdx.x;
  int hh = blk % NH_, c = (blk / NH_) % NC_, b = blk / (NH_ * NC_);
  int tid = threadIdx.x, lane = tid & 31, wv = tid >> 5;   // 8 waves
  size_t obh = ((size_t)b * NH_ + hh) * T_;
  sdt[tid] = dtb[obh + c * CHUNK_ + tid];
  sac[tid] = acb[obh + c * CHUNK_ + tid];
  __syncthreads();
  int half = lane >> 4, l15 = lane & 15;
  int i0 = wv * 32;
  const float* CBb = CB + (size_t)(b * NC_ + c) * CHUNK_ * CHUNK_;
  const unsigned short* xTb = xT + ((size_t)b * NH_ + hh) * HDIM_ * T_;
  f32x8 acc[2][4];
#pragma unroll
  for (int i = 0; i < 2; ++i)
#pragma unroll
    for (int j = 0; j < 4; ++j) acc[i][j] = zero8();
  float ai[2] = {sac[i0 + l15], sac[i0 + 16 + l15]};
  for (int j0 = 0; j0 <= i0; j0 += 32) {           // causal: skip all-zero tiles
    Frag a[2];
#pragma unroll
    for (int fi = 0; fi < 2; ++fi) {
      int i = i0 + fi * 16 + l15;
      const float* cbrow = CBb + (size_t)i * CHUNK_ + j0;
#pragma unroll
      for (int e = 0; e < 16; ++e) {
        int kk = (e < 8) ? (half * 8 + e) : (16 + half * 8 + (e - 8));
        int j = j0 + kk;
        float v = 0.f;
        if (i >= j) v = cbrow[kk] * __expf(ai[fi] - sac[j]) * sdt[j];
        a[fi].u[e] = f2bf(v);
      }
    }
    Frag w[4];
#pragma unroll
    for (int fn = 0; fn < 4; ++fn)
      load_mat_b(w[fn], xTb, T_, fn * 16, c * CHUNK_ + j0, lane, HDIM_ - 1);
#pragma unroll
    for (int fi = 0; fi < 2; ++fi)
#pragma unroll
      for (int fn = 0; fn < 4; ++fn) acc[fi][fn] = wmma_bf16(a[fi], w[fn], acc[fi][fn]);
  }
  float D = Dpar[hh];
#pragma unroll
  for (int fi = 0; fi < 2; ++fi)
#pragma unroll
    for (int fn = 0; fn < 4; ++fn)
#pragma unroll
      for (int r = 0; r < 8; ++r) {
        int i = i0 + fi * 16 + r + 8 * half;
        int p = fn * 16 + l15;
        float xv = bf2f(xTb[(size_t)p * T_ + c * CHUNK_ + i]);
        y[((size_t)b * T_ + c * CHUNK_ + i) * DIN_ + hh * HDIM_ + p] =
            acc[fi][fn][r] + D * xv;
      }
}

// ---------------- SSD: chunk end-states  states[p,n] = (w.*x)^T @ B ----------------
__global__ __launch_bounds__(128) void ssd_states_kernel(
    const unsigned short* __restrict__ xT, const unsigned short* __restrict__ Bt,
    const float* __restrict__ wb, float* __restrict__ states) {
  __shared__ float sw[CHUNK_];
  int blk = blockIdx.x;
  int hh = blk % NH_, c = (blk / NH_) % NC_, b = blk / (NH_ * NC_);
  int tid = threadIdx.x, lane = tid & 31, wv = tid >> 5;   // 4 waves
  size_t obh = ((size_t)b * NH_ + hh) * T_;
  sw[tid] = wb[obh + c * CHUNK_ + tid];
  sw[tid + 128] = wb[obh + c * CHUNK_ + tid + 128];
  __syncthreads();
  int half = lane >> 4, l15 = lane & 15;
  int n0 = wv * 32;
  const unsigned short* xTb = xT + ((size_t)b * NH_ + hh) * HDIM_ * T_;
  const unsigned short* Btb = Bt + (size_t)b * DST_ * T_;
  f32x8 acc[4][2];
#pragma unroll
  for (int i = 0; i < 4; ++i)
#pragma unroll
    for (int j = 0; j < 2; ++j) acc[i][j] = zero8();
  for (int q0 = 0; q0 < CHUNK_; q0 += 32) {
    Frag a[4];
#pragma unroll
    for (int fm = 0; fm < 4; ++fm) {
      int p = fm * 16 + l15;
      const unsigned short* xp = xTb + (size_t)p * T_ + c * CHUNK_ + q0 + half * 8;
      unsigned short t0[8], t1[8];
      *reinterpret_cast<int4*>(t0) = *reinterpret_cast<const int4*>(xp);
      *reinterpret_cast<int4*>(t1) = *reinterpret_cast<const int4*>(xp + 16);
#pragma unroll
      for (int e = 0; e < 8; ++e) {
        a[fm].u[e] = f2bf(bf2f(t0[e]) * sw[q0 + half * 8 + e]);
        a[fm].u[8 + e] = f2bf(bf2f(t1[e]) * sw[q0 + 16 + half * 8 + e]);
      }
    }
    Frag wB[2];
#pragma unroll
    for (int fn = 0; fn < 2; ++fn)
      load_mat_b(wB[fn], Btb, T_, n0 + fn * 16, c * CHUNK_ + q0, lane, DST_ - 1);
#pragma unroll
    for (int fm = 0; fm < 4; ++fm)
#pragma unroll
      for (int fn = 0; fn < 2; ++fn) acc[fm][fn] = wmma_bf16(a[fm], wB[fn], acc[fm][fn]);
  }
  size_t base = (((size_t)(b * NC_ + c) * NH_ + hh) * HDIM_) * DST_;
#pragma unroll
  for (int fm = 0; fm < 4; ++fm)
#pragma unroll
    for (int fn = 0; fn < 2; ++fn)
#pragma unroll
      for (int r = 0; r < 8; ++r) {
        int p = fm * 16 + r + 8 * half;
        int n = n0 + fn * 16 + l15;
        states[base + (size_t)p * DST_ + n] = acc[fm][fn][r];
      }
}

// ---------------- SSD: sequential inter-chunk scan (stores state BEFORE chunk) ------
__global__ void ssd_scan_kernel(const float* __restrict__ states,
                                const float* __restrict__ eb,
                                unsigned short* __restrict__ hpb) {
  int hh = blockIdx.x % NH_, b = blockIdx.x / NH_;
  int tid = threadIdx.x;
  const int PER = (HDIM_ * DST_) / 256;   // 32
  float hp[PER];
#pragma unroll
  for (int k = 0; k < PER; ++k) hp[k] = 0.f;
  size_t obh = ((size_t)b * NH_ + hh) * T_;
  for (int c = 0; c < NC_; ++c) {
    float dec = eb[obh + c * CHUNK_ + CHUNK_ - 1];   // chunk_decay = exp(acum_last)
    size_t base = (((size_t)(b * NC_ + c) * NH_ + hh) * HDIM_) * DST_;
#pragma unroll
    for (int k = 0; k < PER; ++k) {
      int e = tid + k * 256;
      hpb[base + e] = f2bf(hp[k]);                    // hprev before this chunk
      hp[k] = hp[k] * dec + states[base + e];
    }
  }
}

// ---------------- SSD: inter-chunk  y += (C @ hprev^T) .* exp(acum) ----------------
__global__ __launch_bounds__(256) void ssd_inter_kernel(
    const unsigned short* __restrict__ Cn, const unsigned short* __restrict__ hpb,
    const float* __restrict__ eb, float* __restrict__ y) {
  int blk = blockIdx.x;
  int hh = blk % NH_, c = (blk / NH_) % NC_, b = blk / (NH_ * NC_);
  int tid = threadIdx.x, lane = tid & 31, wv = tid >> 5;   // 8 waves
  int half = lane >> 4, l15 = lane & 15;
  int i0 = wv * 32;
  const unsigned short* Cb = Cn + ((size_t)b * T_ + c * CHUNK_) * DST_;
  const unsigned short* hb = hpb + (((size_t)(b * NC_ + c) * NH_ + hh) * HDIM_) * DST_;
  f32x8 acc[2][4];
#pragma unroll
  for (int i = 0; i < 2; ++i)
#pragma unroll
    for (int j = 0; j < 4; ++j) acc[i][j] = zero8();
  for (int n0 = 0; n0 < DST_; n0 += 32) {
    Frag a[2], w[4];
#pragma unroll
    for (int fi = 0; fi < 2; ++fi) load_mat_a(a[fi], Cb, DST_, i0 + fi * 16, n0, lane);
#pragma unroll
    for (int fp = 0; fp < 4; ++fp)
      load_mat_b(w[fp], hb, DST_, fp * 16, n0, lane, HDIM_ - 1);
#pragma unroll
    for (int fi = 0; fi < 2; ++fi)
#pragma unroll
      for (int fp = 0; fp < 4; ++fp) acc[fi][fp] = wmma_bf16(a[fi], w[fp], acc[fi][fp]);
  }
  size_t obh = ((size_t)b * NH_ + hh) * T_;
#pragma unroll
  for (int fi = 0; fi < 2; ++fi)
#pragma unroll
    for (int fp = 0; fp < 4; ++fp)
#pragma unroll
      for (int r = 0; r < 8; ++r) {
        int i = i0 + fi * 16 + r + 8 * half;
        int t = c * CHUNK_ + i;
        int p = fp * 16 + l15;
        float ev = eb[obh + t];
        size_t idx = ((size_t)b * T_ + t) * DIN_ + hh * HDIM_ + p;
        y[idx] += acc[fi][fp][r] * ev;
      }
}

// ---------------- host launcher ----------------
extern "C" void kernel_launch(void* const* d_in, const int* in_sizes, int n_in,
                              void* d_out, int out_size, void* d_ws, size_t ws_size,
                              hipStream_t stream) {
  (void)in_sizes; (void)n_in; (void)out_size; (void)ws_size;
  const int*   x      = (const int*)d_in[0];
  const float* embed  = (const float*)d_in[1];
  const float* pos    = (const float*)d_in[2];
  const float* norm_w = (const float*)d_in[3];
  const float* wi     = (const float*)d_in[4];
  const float* cw     = (const float*)d_in[5];
  const float* cb     = (const float*)d_in[6];
  const float* dtbias = (const float*)d_in[7];
  const float* alog   = (const float*)d_in[8];
  const float* dpar   = (const float*)d_in[9];
  const float* gw     = (const float*)d_in[10];
  const float* wo     = (const float*)d_in[11];
  const float* nfw    = (const float*)d_in[12];
  float* out = (float*)d_out;

  char* ws = (char*)d_ws;
  size_t off = 0;
  auto alloc = [&](size_t bytes) -> char* {
    char* p = ws + off;
    off += (bytes + 255) & ~(size_t)255;
    return p;
  };
  unsigned short* embed_bf = (unsigned short*)alloc((size_t)V_ * DM_ * 2);
  unsigned short* wi_bf    = (unsigned short*)alloc((size_t)NL_ * DPROJ_ * DM_ * 2);
  unsigned short* wo_bf    = (unsigned short*)alloc((size_t)NL_ * DM_ * DIN_ * 2);
  float*          h        = (float*)alloc((size_t)ROWS_ * DM_ * 4);
  unsigned short* u_bf     = (unsigned short*)alloc((size_t)ROWS_ * DM_ * 2);
  float*          zx       = (float*)alloc((size_t)ROWS_ * DPROJ_ * 4);
  unsigned short* xT       = (unsigned short*)alloc((size_t)BT_ * DIN_ * T_ * 2);
  unsigned short* Bn       = (unsigned short*)alloc((size_t)BT_ * T_ * DST_ * 2);
  unsigned short* Bt       = (unsigned short*)alloc((size_t)BT_ * T_ * DST_ * 2);
  unsigned short* Cn       = (unsigned short*)alloc((size_t)BT_ * T_ * DST_ * 2);
  float*          dtb      = (float*)alloc((size_t)BT_ * NH_ * T_ * 4);
  float*          acb      = (float*)alloc((size_t)BT_ * NH_ * T_ * 4);
  float*          wbuf     = (float*)alloc((size_t)BT_ * NH_ * T_ * 4);
  float*          ebuf     = (float*)alloc((size_t)BT_ * NH_ * T_ * 4);
  float*          CBbuf    = (float*)alloc((size_t)BT_ * NC_ * CHUNK_ * CHUNK_ * 4);
  float*          states   = (float*)alloc((size_t)BT_ * NC_ * NH_ * HDIM_ * DST_ * 4);
  unsigned short* hp_bf    = (unsigned short*)alloc((size_t)BT_ * NC_ * NH_ * HDIM_ * DST_ * 2);
  float*          y        = (float*)alloc((size_t)ROWS_ * DIN_ * 4);
  unsigned short* y_bf     = (unsigned short*)alloc((size_t)ROWS_ * DIN_ * 2);
  unsigned short* hf_bf    = (unsigned short*)alloc((size_t)ROWS_ * DM_ * 2);

  // one-time (per call) bf16 weight conversion; bf16 embed (103MB) fits the 192MB L2
  f2bf_kernel<<<2048, 256, 0, stream>>>(embed, embed_bf, (size_t)V_ * DM_);
  f2bf_kernel<<<2048, 256, 0, stream>>>(wi, wi_bf, (size_t)NL_ * DPROJ_ * DM_);
  f2bf_kernel<<<1024, 256, 0, stream>>>(wo, wo_bf, (size_t)NL_ * DM_ * DIN_);

  embed_kernel<<<ROWS_, 256, 0, stream>>>(x, embed, pos, h);

  for (int L = 0; L < NL_; ++L) {
    rmsnorm_kernel<<<ROWS_, 256, 0, stream>>>(h, norm_w + (size_t)L * DM_, u_bf, EPS_RMS_);
    // zxbcdt = u @ Wi^T     [4096 x 4384]
    wmma_gemm_kernel<<<dim3((DPROJ_ + 63) / 64, ROWS_ / 64, 1), 32, 0, stream>>>(
        u_bf, wi_bf + (size_t)L * DPROJ_ * DM_, zx, ROWS_, DPROJ_, DM_, DM_, DM_, DPROJ_,
        0, 0, 0, 0);
    conv_kernel<<<ROWS_, 256, 0, stream>>>(zx, cw + (size_t)L * CONV_ * DC_,
                                           cb + (size_t)L * CONV_, xT, Bn, Bt, Cn);
    dt_kernel<<<BT_ * NH_, 256, 0, stream>>>(zx, dtbias + L * NH_, alog + L * NH_, dtb,
                                             acb, wbuf, ebuf);
    // CB = C @ B^T per chunk (batched over b*c)
    wmma_gemm_kernel<<<dim3(CHUNK_ / 64, CHUNK_ / 64, BT_ * NC_), 32, 0, stream>>>(
        Cn, Bn, CBbuf, CHUNK_, CHUNK_, DST_, DST_, DST_, CHUNK_,
        (long long)CHUNK_ * DST_, (long long)CHUNK_ * DST_, (long long)CHUNK_ * CHUNK_, 0);
    ssd_intra_kernel<<<BT_ * NC_ * NH_, 256, 0, stream>>>(CBbuf, xT, dtb, acb,
                                                          dpar + L * NH_, y);
    ssd_states_kernel<<<BT_ * NC_ * NH_, 128, 0, stream>>>(xT, Bt, wbuf, states);
    ssd_scan_kernel<<<BT_ * NH_, 256, 0, stream>>>(states, ebuf, hp_bf);
    ssd_inter_kernel<<<BT_ * NC_ * NH_, 256, 0, stream>>>(Cn, hp_bf, ebuf, y);
    gated_norm_kernel<<<ROWS_, 256, 0, stream>>>(y, zx, gw + (size_t)L * DIN_, y_bf);
    // h += y @ Wo^T          [4096 x 1024]
    wmma_gemm_kernel<<<dim3(DM_ / 64, ROWS_ / 64, 1), 32, 0, stream>>>(
        y_bf, wo_bf + (size_t)L * DM_ * DIN_, h, ROWS_, DM_, DIN_, DIN_, DIN_, DM_,
        0, 0, 0, 1);
  }

  rmsnorm_kernel<<<ROWS_, 256, 0, stream>>>(h, nfw, hf_bf, EPS_RMS_);
  // logits = hf @ embed^T   [4096 x 50257]   (dominant GEMM, ~422 GFLOP)
  wmma_gemm_kernel<<<dim3((V_ + 63) / 64, ROWS_ / 64, 1), 32, 0, stream>>>(
      hf_bf, embed_bf, out, ROWS_, V_, DM_, DM_, DM_, V_, 0, 0, 0, 0);
}
